// MultiHeadAttention_75625784148083
// MI455X (gfx1250) — matrix operand order repaired and checked by reading, not compiled, not run on hardware
//
#include <hip/hip_runtime.h>

typedef __attribute__((ext_vector_type(16))) _Float16 v16h;
typedef __attribute__((ext_vector_type(8)))  float    v8f;
typedef __attribute__((ext_vector_type(4)))  unsigned int v4u;
typedef __attribute__((ext_vector_type(8)))  int      v8i_;
typedef __attribute__((ext_vector_type(4)))  int      v4i_;

namespace {

constexpr int kD  = 1024;   // model dim
constexpr int kH  = 16;     // heads
constexpr int kHd = 64;     // head dim
constexpr int kS  = 2048;   // seq len
constexpr int kB  = 4;      // batch

__device__ inline v8f vzero8() {
  v8f z = {0.f,0.f,0.f,0.f,0.f,0.f,0.f,0.f};
  return z;
}

__device__ inline v8f wmma_f16(v16h a, v16h b, v8f c) {
  // (neg_a, A, neg_b, B, c_mod, C, reuse_a, reuse_b)
  return __builtin_amdgcn_wmma_f32_16x16x32_f16(false, a, false, b, (short)0, c,
                                                false, false);
}

// ------------------------------------------------------------------
// GEMM: Y[row, col] = sum_k X[row,k] * W[col,k] + bias[col]
//   M = kB*kS = 8192, N = kD = 1024, K = kD = 1024
// OUT_MODE 0: f16 output in [B, H, S, hd] layout (for attention)
// OUT_MODE 1: f32 output in [B*S, D] layout (final result)
// Block tile 128x128, 8 waves each computing a 32x64 sub-tile.
// ------------------------------------------------------------------
template<bool XF16, int OUT_MODE>
__global__ __launch_bounds__(256)
void gemm_kernel(const void* __restrict__ Xv, const float* __restrict__ W,
                 const float* __restrict__ bias, void* __restrict__ Yv)
{
  __shared__ __align__(16) _Float16 Xl[128*32];   // [m][k]
  __shared__ __align__(16) _Float16 Wl[128*32];   // [n][k]

  const int tid  = threadIdx.x;
  const int wave = tid >> 5;
  const int lane = tid & 31;
  const int nl   = lane & 15;
  const int grp  = lane >> 4;          // 0 or 1
  const int m0   = (int)blockIdx.x * 128;
  const int n0   = (int)blockIdx.y * 128;
  const int mofs = (wave >> 1) * 32;
  const int nofs = (wave & 1) * 64;

  v8f acc[2][4];
#pragma unroll
  for (int i = 0; i < 2; ++i)
#pragma unroll
    for (int f = 0; f < 4; ++f) acc[i][f] = vzero8();

  for (int k0 = 0; k0 < kD; k0 += 32) {
    __syncthreads();
    if (XF16) {
      const _Float16* X = (const _Float16*)Xv;
#pragma unroll
      for (int i = 0; i < 2; ++i) {
        const int idx = tid + i*256;            // 512 x uint4
        const int row = idx >> 2, c = (idx & 3) * 8;
        *(uint4*)(&Xl[row*32 + c]) =
            *(const uint4*)(X + (size_t)(m0+row)*kD + k0 + c);
      }
    } else {
      const float* X = (const float*)Xv;
#pragma unroll
      for (int i = 0; i < 4; ++i) {
        const int idx = tid + i*256;            // 1024 x float4
        const int row = idx >> 3, c = (idx & 7) * 4;
        const float4 vv = *(const float4*)(X + (size_t)(m0+row)*kD + k0 + c);
        Xl[row*32 + c + 0] = (_Float16)vv.x;
        Xl[row*32 + c + 1] = (_Float16)vv.y;
        Xl[row*32 + c + 2] = (_Float16)vv.z;
        Xl[row*32 + c + 3] = (_Float16)vv.w;
      }
    }
#pragma unroll
    for (int i = 0; i < 4; ++i) {
      const int idx = tid + i*256;
      const int row = idx >> 3, c = (idx & 7) * 4;
      const float4 vv = *(const float4*)(W + (size_t)(n0+row)*kD + k0 + c);
      Wl[row*32 + c + 0] = (_Float16)vv.x;
      Wl[row*32 + c + 1] = (_Float16)vv.y;
      Wl[row*32 + c + 2] = (_Float16)vv.z;
      Wl[row*32 + c + 3] = (_Float16)vv.w;
    }
    __syncthreads();

    // A fragments (16x32 f16): lanes 0-15 row M=lane, K in {0..7,16..23};
    // lanes 16-31 same rows, K in {8..15,24..31}.
    v16h a[2];
#pragma unroll
    for (int i = 0; i < 2; ++i) {
      const _Float16* p = &Xl[(mofs + i*16 + nl)*32];
      const int kb = grp*8;
      v16h t;
#pragma unroll
      for (int j = 0; j < 8; ++j) t[j]   = p[kb + j];
#pragma unroll
      for (int j = 0; j < 8; ++j) t[8+j] = p[16 + kb + j];
      a[i] = t;
    }
    // B fragments (32x16 f16): lanes 0-15 hold K in {0..7,16..23}, lanes 16-31 K in {8..15,24..31}.
#pragma unroll
    for (int f = 0; f < 4; ++f) {
      const _Float16* p = &Wl[(nofs + f*16 + nl)*32 + grp*8];
      v16h bt;
#pragma unroll
      for (int j = 0; j < 16; ++j) bt[j] = p[(j & 7) + (j >> 3)*16];
#pragma unroll
      for (int i = 0; i < 2; ++i) acc[i][f] = wmma_f16(a[i], bt, acc[i][f]);
    }
  }

  // Epilogue: C/D layout: vgpr r -> row m = r + 8*grp, lane -> col nl.
#pragma unroll
  for (int i = 0; i < 2; ++i) {
    const int mbase = m0 + mofs + i*16 + grp*8;
#pragma unroll
    for (int f = 0; f < 4; ++f) {
      const int col = n0 + nofs + f*16 + nl;
      const float bv = bias[col];
#pragma unroll
      for (int r = 0; r < 8; ++r) {
        const int row = mbase + r;
        const float val = acc[i][f][r] + bv;
        if (OUT_MODE == 0) {
          const int bb = row >> 11, ss = row & (kS - 1);
          const int hh = col >> 6,  dd = col & (kHd - 1);
          ((_Float16*)Yv)[(((size_t)(bb*kH + hh))*kS + ss)*kHd + dd] =
              (_Float16)val;
        } else {
          ((float*)Yv)[(size_t)row*kD + col] = val;
        }
      }
    }
  }
}

// ------------------------------------------------------------------
// Flash attention: one block per (b, h, 128 query rows); 8 waves,
// each wave owns a 16-row Q tile. Keys/values streamed 32 at a time;
// K tiles are DMA'd by the Tensor Data Mover, V tiles transposed by
// hand (TDM cannot transpose; transposed V makes PV B-frags
// contiguous ds_load_b128s).
// ------------------------------------------------------------------
__global__ __launch_bounds__(256)
void attn_kernel(const _Float16* __restrict__ Qg, const _Float16* __restrict__ Kg,
                 const _Float16* __restrict__ Vg, _Float16* __restrict__ Ao)
{
  __shared__ __align__(16) _Float16 Kl[32*64];    // [key][dim]   (TDM target)
  __shared__ __align__(16) _Float16 Vl[64*32];    // [dim][key]   (transposed)
  __shared__ __align__(16) _Float16 Pl[8][16*32]; // per-wave P scratch

  const int tid  = threadIdx.x;
  const int wave = tid >> 5;
  const int lane = tid & 31;
  const int nl   = lane & 15;
  const int grp  = lane >> 4;

  const int qblk = blockIdx.x & 15;          // S/128 = 16
  const int hh   = (blockIdx.x >> 4) & 15;
  const int bb   = blockIdx.x >> 8;

  const size_t headbase = ((size_t)(bb*kH + hh)) * kS * kHd;
  const int qrow0 = qblk*128 + wave*16;

  // Q fragments, pre-scaled by 1/sqrt(hd) * log2(e) so the online softmax
  // can run in base-2 (native v_exp_f32) with zero per-tile multiplies.
  const float cscale = 0.125f * 1.44269504088896f;
  v16h qa[2];
  {
    const _Float16* qp = Qg + headbase + (size_t)(qrow0 + nl)*kHd;
    const int kb = grp*8;
#pragma unroll
    for (int c = 0; c < 2; ++c) {
      v16h t;
#pragma unroll
      for (int j = 0; j < 8; ++j) t[j]   = (_Float16)((float)qp[c*32 + kb + j] * cscale);
#pragma unroll
      for (int j = 0; j < 8; ++j) t[8+j] = (_Float16)((float)qp[c*32 + 16 + kb + j] * cscale);
      qa[c] = t;
    }
  }

  v8f o[4];
#pragma unroll
  for (int f = 0; f < 4; ++f) o[f] = vzero8();
  float rm[8], rs[8];
#pragma unroll
  for (int r = 0; r < 8; ++r) { rm[r] = -3.0e30f; rs[r] = 0.0f; }

  const int key_r = tid >> 3;          // 0..31
  const int key_c = (tid & 7) * 8;     // 0..56
  const unsigned ldsK = (unsigned)(uintptr_t)&Kl[0];  // LDS byte offset

  for (int kt = 0; kt < kS/32; ++kt) {
    const size_t tilebase = headbase + (size_t)kt*32*kHd;
    __syncthreads();

    // ---- K tile via Tensor Data Mover (wave 0 issues the DMA) ----
    if (wave == 0) {
      const unsigned long long ga =
          (unsigned long long)(uintptr_t)(Kg + tilebase);
      v4u g0;
      g0[0] = 1u;                                   // count=1, user mode
      g0[1] = ldsK;                                 // lds_addr
      g0[2] = (unsigned)ga;                         // global_addr[31:0]
      g0[3] = ((unsigned)(ga >> 32) & 0x01FFFFFFu)  // global_addr[56:32]
              | 0x80000000u;                        // type=2 ("image")
      v8i_ g1;
      g1[0] = 0x00010000;        // wg_mask=0, data_size=1 (2 bytes)
      g1[1] = (64 << 16);        // tensor_dim0 = 64 (low16 at bit 48)
      g1[2] = (2048 << 16);      // tensor_dim1 = 2048 (low16 at bit 80)
      g1[3] = (64 << 16);        // tile_dim0 = 64
      g1[4] = 32;                // tile_dim1 = 32, tile_dim2 = 0
      g1[5] = 64;                // tensor_dim0_stride = 64
      g1[6] = 0;
      g1[7] = 0;
      v4i_ g2 = {0,0,0,0};       // 2D tensor: groups 2/3 unused
      v4i_ g3 = {0,0,0,0};
      v8i_ g4 = {0,0,0,0,0,0,0,0};
      __builtin_amdgcn_tensor_load_to_lds(g0, g1, g2, g3, g4, 0);
    }

    // ---- V tile: manual transpose into [dim][key] ----
    {
      union { uint4 u; _Float16 hx[8]; } uu;
      uu.u = *(const uint4*)(Vg + tilebase + (size_t)key_r*kHd + key_c);
#pragma unroll
      for (int j = 0; j < 8; ++j) Vl[(key_c + j)*32 + key_r] = uu.hx[j];
    }
    if (kt + 1 < kS/32) {
      __builtin_prefetch((const void*)(Vg + tilebase + 32*kHd + tid*8), 0, 1);
    }
    __builtin_amdgcn_s_wait_tensorcnt(0);   // no-op for waves 1..7
    __syncthreads();

    // Scores: Q(16x64) . K^T for two 16-key tiles.
    v8f s0 = vzero8(), s1 = vzero8();
#pragma unroll
    for (int c = 0; c < 2; ++c) {
      const _Float16* p0 = &Kl[(size_t)nl*kHd      + c*32 + grp*8];
      const _Float16* p1 = &Kl[(size_t)(16+nl)*kHd + c*32 + grp*8];
      v16h b0, b1;
#pragma unroll
      for (int j = 0; j < 16; ++j) { b0[j] = p0[(j & 7) + (j >> 3)*16]; b1[j] = p1[(j & 7) + (j >> 3)*16]; }
      s0 = wmma_f16(qa[c], b0, s0);
      s1 = wmma_f16(qa[c], b1, s1);
    }

    // Online softmax in base-2. Row m = r + 8*grp spans 16 lanes (cols).
#pragma unroll
    for (int r = 0; r < 8; ++r) {
      const float a0 = s0[r];
      const float a1 = s1[r];
      float mx = fmaxf(a0, a1);
#pragma unroll
      for (int d = 1; d < 16; d <<= 1) mx = fmaxf(mx, __shfl_xor(mx, d, 32));
      const float mnew = fmaxf(rm[r], mx);
      const float corr = exp2f(rm[r] - mnew);
      rm[r] = mnew;
      const float p0 = exp2f(a0 - mnew);
      const float p1 = exp2f(a1 - mnew);
      float ts = p0 + p1;
#pragma unroll
      for (int d = 1; d < 16; d <<= 1) ts += __shfl_xor(ts, d, 32);
      rs[r] = rs[r]*corr + ts;
#pragma unroll
      for (int f = 0; f < 4; ++f) o[f][r] *= corr;
      // C-layout -> LDS (row-major 16x32) for A-frag reload
      _Float16* pp = &Pl[wave][(r + 8*grp)*32];
      pp[nl]      = (_Float16)p0;
      pp[16 + nl] = (_Float16)p1;
    }

    // Reload P as A-fragment (16x32)
    v16h pa;
    {
      const _Float16* p = &Pl[wave][nl*32];
      const int kb = grp*8;
#pragma unroll
      for (int j = 0; j < 8; ++j) pa[j]   = p[kb + j];
#pragma unroll
      for (int j = 0; j < 8; ++j) pa[8+j] = p[16 + kb + j];
    }
    // O(16x64) += P(16x32) . V(32x64)
#pragma unroll
    for (int f = 0; f < 4; ++f) {
      const _Float16* p = &Vl[(f*16 + nl)*32 + grp*8];
      v16h bv;
#pragma unroll
      for (int j = 0; j < 16; ++j) bv[j] = p[(j & 7) + (j >> 3)*16];
      o[f] = wmma_f16(pa, bv, o[f]);
    }
  }

  // Normalize and store f16 into [B, S, D] (heads re-merged).
#pragma unroll
  for (int r = 0; r < 8; ++r) {
    const float inv = 1.0f / rs[r];
#pragma unroll
    for (int f = 0; f < 4; ++f) o[f][r] *= inv;
  }
#pragma unroll
  for (int f = 0; f < 4; ++f) {
    const int col = hh*kHd + f*16 + nl;
#pragma unroll
    for (int r = 0; r < 8; ++r) {
      const int row = qrow0 + r + 8*grp;
      Ao[((size_t)bb*kS + row)*kD + col] = (_Float16)o[f][r];
    }
  }
}

} // anonymous namespace

extern "C" void kernel_launch(void* const* d_in, const int* in_sizes, int n_in,
                              void* d_out, int out_size, void* d_ws, size_t ws_size,
                              hipStream_t stream)
{
  const float* q  = (const float*)d_in[0];
  const float* k  = (const float*)d_in[1];
  const float* v  = (const float*)d_in[2];
  const float* Wq = (const float*)d_in[3];
  const float* bq = (const float*)d_in[4];
  const float* Wk = (const float*)d_in[5];
  const float* bk = (const float*)d_in[6];
  const float* Wv = (const float*)d_in[7];
  const float* bv = (const float*)d_in[8];
  const float* Wo = (const float*)d_in[9];
  const float* bo = (const float*)d_in[10];

  const size_t elems = (size_t)kB * kS * kD;     // 8Mi elements
  _Float16* qf = (_Float16*)d_ws;                // [B,H,S,hd] f16
  _Float16* kf = qf + elems;
  _Float16* vf = kf + elems;
  _Float16* af = vf + elems;                     // [B,S,D] f16
  (void)in_sizes; (void)n_in; (void)out_size; (void)ws_size;

  dim3 ggrid(kB*kS/128, kD/128);                 // 64 x 8
  dim3 blk(256);

  gemm_kernel<false, 0><<<ggrid, blk, 0, stream>>>((const void*)q, Wq, bq, (void*)qf);
  gemm_kernel<false, 0><<<ggrid, blk, 0, stream>>>((const void*)k, Wk, bk, (void*)kf);
  gemm_kernel<false, 0><<<ggrid, blk, 0, stream>>>((const void*)v, Wv, bv, (void*)vf);

  attn_kernel<<<kB*kH*(kS/128), blk, 0, stream>>>(qf, kf, vf, af);

  gemm_kernel<true, 1><<<ggrid, blk, 0, stream>>>((const void*)af, Wo, bo, d_out);
}
